// QuerySelector_15015205667050
// MI455X (gfx1250) — matrix operand order repaired and checked by reading, not compiled, not run on hardware
//
#include <hip/hip_runtime.h>
#include <hip/hip_bf16.h>
#include <float.h>

// Problem constants (B, N, C, H) = (256, 197, 768, 12), hd = 64
#define BB     256
#define NN     197
#define CC     768
#define HH     12
#define HD     64
#define MROWS  (BB * NN)      // 50432 tokens
#define NT     196            // tokens after dropping cls
#define KSTEPS (CC / 32)      // 24 K-steps of 32 (bf16 WMMA K)

// GEMM tiling
#define MT      64                  // rows per block  (50432 = 788 * 64)
#define GWAVES  16                  // 512 threads
#define ROWPAD  80                  // LDS bytes per B row: 64 data + 16 pad (bank-conflict-free)
#define BCHUNK  (CC * ROWPAD)       // 61440 bytes per staged K-chunk

typedef __attribute__((ext_vector_type(16))) __bf16 v16bf;
typedef __attribute__((ext_vector_type(8)))  float  v8f;
typedef __attribute__((ext_vector_type(4)))  float  v4f;

union Frag16 { v16bf v; uint4 q[2]; };

__device__ __forceinline__ unsigned short f2bf(float f) {
  unsigned int u = __float_as_uint(f);
  u += 0x7FFFu + ((u >> 16) & 1u);   // round-to-nearest-even
  return (unsigned short)(u >> 16);
}

// CDNA5 async global->LDS copy (ASYNCcnt path, CDNA5 ISA).
// First operand VGPR = LDS byte address (low 32 bits of flat shared pointer),
// second = 64-bit global address, GV mode (SADDR=off).
__device__ __forceinline__ void async_b128(unsigned lds_off, const void* g) {
  asm volatile("global_load_async_to_lds_b128 %0, %1, off"
               :: "v"(lds_off), "v"(g) : "memory");
}
__device__ __forceinline__ void wait_asynccnt0() {
  asm volatile("s_wait_asynccnt 0x0" ::: "memory");
}

// ---------------------------------------------------------------- converts
__global__ void k_f32_to_bf16_v4(const float* __restrict__ src,
                                 unsigned short* __restrict__ dst, int n4) {
  int i = blockIdx.x * blockDim.x + threadIdx.x;
  if (i >= n4) return;
  float4 f = ((const float4*)src)[i];
  ushort4 o;
  o.x = f2bf(f.x); o.y = f2bf(f.y); o.z = f2bf(f.z); o.w = f2bf(f.w);
  ((ushort4*)dst)[i] = o;
}

// ---------------------------------------------------------------- q0 = x[:,0,:] @ Wq^T   (B x C)
__global__ void k_q0(const float* __restrict__ x, const float* __restrict__ Wqkv,
                     float* __restrict__ q0) {
  int gid = blockIdx.x * blockDim.x + threadIdx.x;
  if (gid >= BB * CC) return;
  int b = gid / CC, j = gid - b * CC;
  const float4* xr = (const float4*)(x + (size_t)b * NN * CC);      // token 0
  const float4* wr = (const float4*)(Wqkv + (size_t)j * CC);        // Wq rows = first C rows
  float s = 0.f;
  for (int c = 0; c < CC / 4; ++c) {
    float4 a = xr[c]; float4 w = wr[c];
    s = fmaf(a.x, w.x, fmaf(a.y, w.y, fmaf(a.z, w.z, fmaf(a.w, w.w, s))));
  }
  q0[gid] = s;
}

// ---------------------------------------------------------------- w_eff[b,h,c] = scale * sum_d q0[b,h,d] * Wk[h*64+d, c]
__global__ void k_weff(const float* __restrict__ q0, const float* __restrict__ Wqkv,
                       float* __restrict__ weff) {
  int gid = blockIdx.x * blockDim.x + threadIdx.x;
  if (gid >= BB * HH * CC) return;
  int c  = gid % CC;
  int bh = gid / CC;
  int h  = bh % HH;
  int b  = bh / HH;
  const float* q = q0 + (size_t)b * CC + h * HD;
  const float* w = Wqkv + (size_t)(CC + h * HD) * CC + c;           // Wk block
  float s = 0.f;
  #pragma unroll 8
  for (int d = 0; d < HD; ++d) s = fmaf(q[d], w[(size_t)d * CC], s);
  weff[gid] = s * 0.125f;                                           // hd^-0.5
}

// ---------------------------------------------------------------- logits[b,h,m] = w_eff[b,h,:] . x[b,m,:]
__global__ void k_logits(const float* __restrict__ x, const float* __restrict__ weff,
                         float* __restrict__ logits) {
  int bh = blockIdx.x;
  int m  = threadIdx.x;
  if (m >= NN) return;
  int b = bh / HH;
  const float4* w  = (const float4*)(weff + (size_t)bh * CC);
  const float4* xr = (const float4*)(x + ((size_t)b * NN + m) * CC);
  float s = 0.f;
  for (int c = 0; c < CC / 4; ++c) {
    float4 a = xr[c]; float4 ww = w[c];
    s = fmaf(a.x, ww.x, fmaf(a.y, ww.y, fmaf(a.z, ww.z, fmaf(a.w, ww.w, s))));
  }
  logits[(size_t)bh * NN + m] = s;
}

// ---------------------------------------------------------------- ||v||: bf16 WMMA GEMM, async-LDS staged B
// Block: 64 token rows, 512 threads / 16 waves. Wave owns 3 col-tiles x 4 row-tiles.
// B (Wv) K-chunks are DMA'd to LDS (double-buffered) with global_load_async_to_lds_b128.
__global__ __launch_bounds__(512)
void k_vnorm_gemm(const unsigned short* __restrict__ xbf,   // [MROWS][CC] bf16
                  const unsigned short* __restrict__ wvbf,  // [CC out][CC k] bf16 (Wv row-major)
                  float* __restrict__ vnorm) {
  extern __shared__ char smem[];        // 2 * BCHUNK bytes, dynamic
  __shared__ float lsum[MT];
  const int tid   = threadIdx.x;
  const int wave  = tid >> 5;
  const int lane  = tid & 31;
  const int half  = lane >> 4;          // 0: lanes 0-15, 1: lanes 16-31
  const int l16   = lane & 15;
  const int mbase = blockIdx.x * MT;
  const unsigned smem_off = (unsigned)(size_t)smem;   // flat LDS addr: low 32b = LDS offset

  if (tid < MT) lsum[tid] = 0.f;

  // Stage K-chunk 0 into buffer 0: 768 rows x 64B = 3072 x b128, 6 per thread.
  #pragma unroll
  for (int j = 0; j < 6; ++j) {
    int c = tid + j * 512;
    int n = c >> 2, q = c & 3;
    async_b128(smem_off + (unsigned)(n * ROWPAD + q * 16),
               wvbf + (size_t)n * CC + q * 8);
  }

  v8f zero = {0.f, 0.f, 0.f, 0.f, 0.f, 0.f, 0.f, 0.f};
  v8f acc[3][4];
  #pragma unroll
  for (int i = 0; i < 3; ++i)
    #pragma unroll
    for (int rt = 0; rt < 4; ++rt) acc[i][rt] = zero;

  // A fragment row bases (ISA 16-bit A 16x32 layout):
  //  lanes 0-15  -> elems 0..7 = K 0..7,  elems 8..15 = K 16..23
  //  lanes 16-31 -> elems 0..7 = K 8..15, elems 8..15 = K 24..31
  const unsigned short* aptr[4];
  #pragma unroll
  for (int rt = 0; rt < 4; ++rt)
    aptr[rt] = xbf + (size_t)(mbase + rt * 16 + l16) * CC + half * 8;

  wait_asynccnt0();
  __syncthreads();

  for (int ks = 0; ks < KSTEPS; ++ks) {
    const int kbase = ks * 32;
    // Kick DMA of next K-chunk into the other buffer while we compute this one.
    if (ks + 1 < KSTEPS) {
      const unsigned nxt = smem_off + (unsigned)(((ks + 1) & 1) * BCHUNK);
      const unsigned short* gk = wvbf + (ks + 1) * 32;
      #pragma unroll
      for (int j = 0; j < 6; ++j) {
        int c = tid + j * 512;
        int n = c >> 2, q = c & 3;
        async_b128(nxt + (unsigned)(n * ROWPAD + q * 16),
                   gk + (size_t)n * CC + q * 8);
      }
      __builtin_prefetch(aptr[0] + kbase + 32, 0, 1);   // global_prefetch_b8 on A stream
    }

    Frag16 a[4];
    #pragma unroll
    for (int rt = 0; rt < 4; ++rt) {
      a[rt].q[0] = *(const uint4*)(aptr[rt] + kbase);
      a[rt].q[1] = *(const uint4*)(aptr[rt] + kbase + 16);
    }

    const char* bufc = smem + (ks & 1) * BCHUNK;
    #pragma unroll
    for (int i = 0; i < 3; ++i) {
      int ct = wave + i * GWAVES;                       // 0..47
      // B layout: lanes 0-15 hold K 0..7,16..23, lanes 16-31 hold K 8..15,24..31, col = ct*16 + l16
      const char* bp = bufc + (ct * 16 + l16) * ROWPAD + half * 16;
      Frag16 b;
      b.q[0] = ((const uint4*)bp)[0];
      b.q[1] = ((const uint4*)bp)[2];
      #pragma unroll
      for (int rt = 0; rt < 4; ++rt)
        acc[i][rt] = __builtin_amdgcn_wmma_f32_16x16x32_bf16(
            false, a[rt].v, false, b.v, (short)0, acc[i][rt], false, false);
    }

    if (ks + 1 < KSTEPS) wait_asynccnt0();              // next chunk landed
    __syncthreads();                                    // everyone done reading cur buffer
  }

  // Square & reduce: C/D layout -> lane half h, VGPR j holds row (rt*16 + 8h + j), col l16.
  #pragma unroll
  for (int rt = 0; rt < 4; ++rt)
    #pragma unroll
    for (int j = 0; j < 8; ++j) {
      float s = 0.f;
      #pragma unroll
      for (int i = 0; i < 3; ++i) { float e = acc[i][rt][j]; s = fmaf(e, e, s); }
      #pragma unroll
      for (int o = 1; o < 16; o <<= 1) s += __shfl_xor(s, o, 32);   // 16-lane half reduce
      if (l16 == 0) atomicAdd(&lsum[rt * 16 + half * 8 + j], s);
    }
  __syncthreads();
  if (tid < MT) vnorm[mbase + tid] = sqrtf(lsum[tid]);
}

// ---------------------------------------------------------------- per-batch: softmax score, argsort, cdf, token pick
__global__ __launch_bounds__(256)
void k_score_select(const float* __restrict__ logits, const float* __restrict__ vnorm,
                    const float* __restrict__ ys, int* __restrict__ gidx) {
  __shared__ float l[HH * NN];
  __shared__ float red[256];
  __shared__ float hmax[HH], hsum[HH];
  __shared__ float key[256];
  __shared__ int   sidx[256];
  __shared__ float sa[256], sb[256];

  const int b = blockIdx.x, tid = threadIdx.x;
  for (int i = tid; i < HH * NN; i += 256) l[i] = logits[(size_t)b * HH * NN + i];
  __syncthreads();

  for (int h = 0; h < HH; ++h) {
    float v = -FLT_MAX;
    for (int m = tid; m < NN; m += 256) v = fmaxf(v, l[h * NN + m]);
    red[tid] = v; __syncthreads();
    for (int o = 128; o > 0; o >>= 1) { if (tid < o) red[tid] = fmaxf(red[tid], red[tid + o]); __syncthreads(); }
    if (tid == 0) hmax[h] = red[0];
    __syncthreads();
    float s = 0.f;
    for (int m = tid; m < NN; m += 256) s += __expf(l[h * NN + m] - hmax[h]);
    red[tid] = s; __syncthreads();
    for (int o = 128; o > 0; o >>= 1) { if (tid < o) red[tid] += red[tid + o]; __syncthreads(); }
    if (tid == 0) hsum[h] = red[0];
    __syncthreads();
  }

  // score[m] = (sum_h attn[b,h,0,m]) * ||v[b,m]||, m = 1..196.
  // (Reference's score normalization cancels inside norm_cdf; skip it.)
  float sc = FLT_MAX;                                   // padding sorts to the end
  if (tid < NT) {
    int t = tid + 1;
    float a = 0.f;
    #pragma unroll
    for (int h = 0; h < HH; ++h) a += __expf(l[h * NN + t] - hmax[h]) / hsum[h];
    sc = a * vnorm[(size_t)b * NN + t];
  }
  key[tid] = sc; sidx[tid] = tid;
  __syncthreads();

  // Bitonic sort, ascending, 256 wide
  for (int k = 2; k <= 256; k <<= 1)
    for (int j = k >> 1; j > 0; j >>= 1) {
      int p = tid ^ j;
      if (p > tid) {
        bool up = ((tid & k) == 0);
        float ka = key[tid], kb = key[p];
        if ((ka > kb) == up) {
          key[tid] = kb; key[p] = ka;
          int ia = sidx[tid]; sidx[tid] = sidx[p]; sidx[p] = ia;
        }
      }
      __syncthreads();
    }

  // Inclusive prefix sum (Hillis-Steele) of sorted scores
  sa[tid] = (tid < NT) ? key[tid] : 0.f;
  __syncthreads();
  float* src = sa; float* dst = sb;
  for (int off = 1; off < 256; off <<= 1) {
    float v = src[tid];
    if (tid >= off) v += src[tid - off];
    dst[tid] = v;
    __syncthreads();
    float* tmp = src; src = dst; dst = tmp;
  }
  float cmin = src[0], cmax = src[NT - 1];
  float inv  = 1.f / (cmax - cmin);
  if (tid < NT) dst[tid] = (src[tid] - cmin) * inv;     // norm_cdf
  __syncthreads();

  if (tid < NT) {
    float y = ys[tid];
    float best = fabsf(y - dst[0]); int bk = 0;
    for (int k = 1; k < NT; ++k) {
      float d = fabsf(y - dst[k]);
      if (d < best) { best = d; bk = k; }               // first-min tie-break == argmin
    }
    gidx[b * NT + tid] = 1 + sidx[bk];                  // row into raw_x[b, :, :]
  }
}

// ---------------------------------------------------------------- output assembly (non-temporal streaming)
__global__ void k_copy4_nt(const float* __restrict__ src, float* __restrict__ dst, int n4) {
  int i = blockIdx.x * blockDim.x + threadIdx.x;
  if (i >= n4) return;
  v4f v = __builtin_nontemporal_load(((const v4f*)src) + i);
  __builtin_nontemporal_store(v, ((v4f*)dst) + i);
}
__global__ void k_copy1_nt(const float* __restrict__ src, float* __restrict__ dst, int n) {
  int i = blockIdx.x * blockDim.x + threadIdx.x;
  if (i >= n) return;
  float v = __builtin_nontemporal_load(src + i);
  __builtin_nontemporal_store(v, dst + i);
}
__global__ void k_select_nt(const float* __restrict__ raw_x, const int* __restrict__ gidx,
                            float* __restrict__ outsel) {
  int i = blockIdx.x * blockDim.x + threadIdx.x;        // over B*NN*CC/4
  const int n4 = BB * NN * (CC / 4);
  if (i >= n4) return;
  int c4  = i % (CC / 4);
  int row = i / (CC / 4);
  int n   = row % NN;
  int b   = row / NN;
  int srcn = (n == 0) ? 0 : gidx[b * NT + (n - 1)];
  const v4f* s = (const v4f*)(raw_x + ((size_t)b * NN + srcn) * CC);
  v4f v = __builtin_nontemporal_load(s + c4);
  __builtin_nontemporal_store(v, ((v4f*)outsel) + i);
}

// ---------------------------------------------------------------- launcher
extern "C" void kernel_launch(void* const* d_in, const int* in_sizes, int n_in,
                              void* d_out, int out_size, void* d_ws, size_t ws_size,
                              hipStream_t stream) {
  (void)in_sizes; (void)n_in; (void)out_size; (void)ws_size;
  const float* x      = (const float*)d_in[0];
  // d_in[1] = n (197), compile-time constant here
  const float* raw_x  = (const float*)d_in[2];
  const float* policy = (const float*)d_in[3];
  const float* Wqkv   = (const float*)d_in[4];
  const float* ys     = (const float*)d_in[5];
  float* out = (float*)d_out;

  // Workspace carve (256B aligned): ~92 MB total
  char*  ws  = (char*)d_ws;
  size_t off = 0;
  auto carve = [&](size_t bytes) -> char* {
    off = (off + 255) & ~(size_t)255;
    char* p = ws + off;
    off += bytes;
    return p;
  };
  unsigned short* xbf  = (unsigned short*)carve((size_t)MROWS * CC * 2);
  unsigned short* wvbf = (unsigned short*)carve((size_t)CC * CC * 2);
  float* q0    = (float*)carve((size_t)BB * CC * 4);
  float* weff  = (float*)carve((size_t)BB * HH * CC * 4);
  float* logit = (float*)carve((size_t)BB * HH * NN * 4);
  float* vnorm = (float*)carve((size_t)MROWS * 4);
  int*   gidx  = (int*)carve((size_t)BB * NT * 4);

  const int n4x = MROWS * CC / 4;                       // 9,682,944 (divisible by 256)
  const int n4w = CC * CC / 4;                          // 147,456

  k_f32_to_bf16_v4<<<n4x / 256, 256, 0, stream>>>(x, xbf, n4x);
  k_f32_to_bf16_v4<<<n4w / 256, 256, 0, stream>>>(Wqkv + (size_t)2 * CC * CC, wvbf, n4w);
  k_q0<<<BB * CC / 256, 256, 0, stream>>>(x, Wqkv, q0);
  k_weff<<<BB * HH * CC / 256, 256, 0, stream>>>(q0, Wqkv, weff);
  k_logits<<<BB * HH, 224, 0, stream>>>(x, weff, logit);
  k_vnorm_gemm<<<MROWS / MT, 512, 2 * BCHUNK, stream>>>(xbf, wvbf, vnorm);
  k_score_select<<<BB, 256, 0, stream>>>(logit, vnorm, ys, gidx);

  k_copy4_nt<<<n4x / 256, 256, 0, stream>>>(x, out, n4x);
  k_select_nt<<<n4x / 256, 256, 0, stream>>>(raw_x, gidx, out + (size_t)MROWS * CC);
  k_copy1_nt<<<(MROWS + 255) / 256, 256, 0, stream>>>(policy, out + (size_t)2 * MROWS * CC, MROWS);
}
